// IPSNet_5738076307943
// MI455X (gfx1250) — matrix operand order repaired and checked by reading, not compiled, not run on hardware
//
#include <hip/hip_runtime.h>
#include <hip/hip_bf16.h>

// ---- problem constants (from reference) ----
#define BB    8
#define NN    16384
#define CC    512
#define DD    128
#define MMEM  1024
#define IITER 1024
#define NITER 15          // ceil((N-M)/I)
#define CAND  2048        // M + I candidates per selection round
#define MASKK 64
#define EPSF  1e-5f
#define KCH   (CC / 64)   // 8 K-chunks of 64

typedef __bf16 bf16;
typedef __attribute__((ext_vector_type(16))) __bf16 v16bf;
typedef __attribute__((ext_vector_type(8)))  float  v8f;

typedef bf16 ATile[8][2][32][16];   // [rowgroup][kk][lane][h]  16 KB
typedef bf16 BTile[2][8][32][16];   // [kk][nt][lane][h]        16 KB

__device__ __forceinline__ bf16 f2bf(float f) {
  unsigned u = __builtin_bit_cast(unsigned, f);
  u += 0x7FFFu + ((u >> 16) & 1u);            // round-to-nearest-even
  unsigned short h = (unsigned short)(u >> 16);
  return __builtin_bit_cast(bf16, h);
}

__device__ __forceinline__ unsigned pack2(bf16 a, bf16 b) {
  unsigned short ua = __builtin_bit_cast(unsigned short, a);
  unsigned short ub = __builtin_bit_cast(unsigned short, b);
  return (unsigned)ua | ((unsigned)ub << 16);
}

// Stage one K=64 chunk into WMMA-fragment-layout LDS tiles (ISA 7.12.2).
__device__ __forceinline__ void stage_tile(int kc, ATile& A, BTile& Bt,
                                           const float* __restrict__ patches,
                                           const float* __restrict__ W_enc,
                                           size_t tokBase,
                                           const float* s_mu, const float* s_rs,
                                           int tid) {
  // activations: float2 along K, normalize, pack bf16x2, scatter into
  // A-frag layout: koff(h,lane) = 2p + 8*[p>=4] + 8*[lane>=16] + q, h=2p+q
  for (int e = tid; e < 128 * 32; e += 256) {
    int row = e >> 5;                 // 0..127
    int kl  = (e & 31) * 2;           // even k within 64-chunk
    const float* gp = patches + (tokBase + row) * CC + kc * 64 + kl;
    float2 v = *(const float2*)gp;
    float mu = s_mu[row], rs = s_rs[row];
    unsigned pk = pack2(f2bf((v.x - mu) * rs), f2bf((v.y - mu) * rs));
    int kk = kl >> 5;
    int l32 = kl & 31;
    int laneHi = (l32 >> 3) & 1;      // which 16-lane half holds this K
    int lp = l32 - 8 * laneHi;        // in {0,2,4,6,16,18,20,22}
    int p  = ((lp & 7) >> 1) + ((lp >= 16) ? 4 : 0);
    *(unsigned*)&A[row >> 4][kk][laneHi * 16 + (row & 15)][2 * p] = pk;
  }
  // W_enc: paired-K reads (d-coalesced), B-frag layout:
  // lanes 0-15 hold K 0..7,16..23; lanes 16-31 hold K 8..15,24..31
  for (int e = tid; e < 32 * 128; e += 256) {
    int kp = e >> 7;                  // K pair 0..31
    int d  = e & 127;
    int k0 = kc * 64 + kp * 2;
    float w0 = W_enc[(size_t)k0 * DD + d];
    float w1 = W_enc[(size_t)(k0 + 1) * DD + d];
    unsigned pk = pack2(f2bf(w0), f2bf(w1));
    int kl  = kp * 2;
    int kk  = kl >> 5;
    int klo = kl & 31;
    int laneHi = (klo >> 3) & 1;
    int h      = (klo & 7) + 8 * (klo >> 4);  // even
    *(unsigned*)&Bt[kk][d >> 4][laneHi * 16 + (d & 15)][h] = pk;
  }
}

// =====================================================================
// Kernel 1: fused LayerNorm -> bf16 WMMA GEMM (x@W_enc) -> BN -> ReLU
//           -> dot(w_score)  => one scalar score per token.
// 256 threads (8 waves); 128 tokens/block; K tiled by 64 with ping-pong
// LDS tiles so global staging of chunk kc+1 overlaps WMMAs of chunk kc.
// B fragments are register double-buffered so ds loads overlap WMMA.
// =====================================================================
__global__ __launch_bounds__(256)
void encode_score_kernel(const float* __restrict__ patches,
                         const float* __restrict__ W_enc,
                         const float* __restrict__ b_enc,
                         const float* __restrict__ bn_gamma,
                         const float* __restrict__ bn_beta,
                         const float* __restrict__ bn_mean,
                         const float* __restrict__ bn_var,
                         const float* __restrict__ w_score,
                         float* __restrict__ r_out) {
  __shared__ alignas(32) ATile laA[2];   // ping-pong, 32 KB
  __shared__ alignas(32) BTile lbB[2];   // ping-pong, 32 KB
  __shared__ float s_mu[128], s_rs[128];
  __shared__ float s_scale[DD], s_shift[DD], s_wsc[DD];

  const int tid  = threadIdx.x;
  const int lane = tid & 31;
  const int wv   = tid >> 5;
  const size_t tokBase = (size_t)blockIdx.x * 128;

  // Fold BN + bias: h = acc*scale + shift, shift absorbs b_enc.
  if (tid < DD) {
    float sc = bn_gamma[tid] * rsqrtf(bn_var[tid] + EPSF);
    s_scale[tid] = sc;
    s_shift[tid] = bn_beta[tid] - bn_mean[tid] * sc + b_enc[tid] * sc;
    s_wsc[tid]   = w_score[tid];
  }

  // LayerNorm stats: wave wv owns rows [wv*16, wv*16+16); 32 lanes x 16 floats.
  for (int rr = 0; rr < 16; ++rr) {
    int row = wv * 16 + rr;
    const float4* p4 = (const float4*)(patches + (tokBase + row) * CC) + lane * 4;
    float sum = 0.f, sq = 0.f;
#pragma unroll
    for (int q = 0; q < 4; ++q) {
      float4 v = p4[q];
      sum += v.x + v.y + v.z + v.w;
      sq  += v.x*v.x + v.y*v.y + v.z*v.z + v.w*v.w;
    }
#pragma unroll
    for (int m = 1; m < 32; m <<= 1) {
      sum += __shfl_xor(sum, m, 32);
      sq  += __shfl_xor(sq,  m, 32);
    }
    if (lane == 0) {
      float mu  = sum * (1.f / CC);
      float var = sq * (1.f / CC) - mu * mu;
      s_mu[row] = mu;
      s_rs[row] = rsqrtf(var + EPSF);
    }
  }
  __syncthreads();

  v8f acc[8];
#pragma unroll
  for (int i = 0; i < 8; ++i)
    acc[i] = (v8f){0.f,0.f,0.f,0.f,0.f,0.f,0.f,0.f};

  // prologue: stage chunk 0
  stage_tile(0, laA[0], lbB[0], patches, W_enc, tokBase, s_mu, s_rs, tid);
  __syncthreads();

  for (int kc = 0; kc < KCH; ++kc) {
    const int cur = kc & 1;
    // issue next chunk's global loads first; they complete under the WMMAs
    if (kc + 1 < KCH)
      stage_tile(kc + 1, laA[cur ^ 1], lbB[cur ^ 1],
                 patches, W_enc, tokBase, s_mu, s_rs, tid);

    const ATile& A  = laA[cur];
    const BTile& Bt = lbB[cur];
#pragma unroll
    for (int kk = 0; kk < 2; ++kk) {
      v16bf afrag = *(const v16bf*)(&A[wv][kk][lane][0]);
      v16bf bcur  = *(const v16bf*)(&Bt[kk][0][lane][0]);
#pragma unroll
      for (int nt = 0; nt < 8; ++nt) {
        v16bf bnext = bcur;
        if (nt < 7) bnext = *(const v16bf*)(&Bt[kk][nt + 1][lane][0]);
        acc[nt] = __builtin_amdgcn_wmma_f32_16x16x32_bf16(
            false, afrag, false, bcur, (short)0, acc[nt], false, false);
        bcur = bnext;
      }
    }
    // one barrier per chunk: next tile fully staged AND current tile fully read
    __syncthreads();
  }

  // Epilogue: BN + ReLU + dot(w_score), reduced across the 16 columns/lane-group.
  // C/D layout: lane = col (mod 16), vgpr v = row v (+8 for lanes 16-31).
  float ps[8];
#pragma unroll
  for (int v = 0; v < 8; ++v) ps[v] = 0.f;
#pragma unroll
  for (int nt = 0; nt < 8; ++nt) {
    int col = nt * 16 + (lane & 15);
    float scale = s_scale[col], shift = s_shift[col], wsc = s_wsc[col];
#pragma unroll
    for (int v = 0; v < 8; ++v) {
      float h = acc[nt][v] * scale + shift;
      ps[v] += fmaxf(h, 0.f) * wsc;
    }
  }
#pragma unroll
  for (int v = 0; v < 8; ++v) {
    float s = ps[v];
#pragma unroll
    for (int m = 1; m < 16; m <<= 1) s += __shfl_xor(s, m, 32);
    if (lane == 0)       r_out[tokBase + wv * 16 + v]     = s;
    else if (lane == 16) r_out[tokBase + wv * 16 + 8 + v] = s;
  }
}

// =====================================================================
// Kernel 2: 15 serial selection rounds per batch, all in LDS.
// One 1024-thread block per batch element. Raw-score domain top-M
// (softmax is monotonic); knocked-out entries -> -inf -> fall out.
// =====================================================================
__device__ __forceinline__ float hash01(unsigned x) {
  x ^= x >> 16; x *= 0x7feb352du; x ^= x >> 15; x *= 0x846ca68bu; x ^= x >> 16;
  return (float)(x >> 8) * (1.0f / 16777216.0f);
}

__device__ __forceinline__ void bitonic_sort_desc(float* sc, int* id, int tid) {
  for (int k = 2; k <= CAND; k <<= 1)
    for (int j = k >> 1; j > 0; j >>= 1) {
      __syncthreads();
      int mask = j - 1;
      int a = ((tid & ~mask) << 1) | (tid & mask);
      int c = a | j;
      bool desc = ((a & k) == 0);
      float sa = sc[a], sb = sc[c];
      if (desc ? (sa < sb) : (sa > sb)) {
        sc[a] = sb; sc[c] = sa;
        int t2 = id[a]; id[a] = id[c]; id[c] = t2;
      }
    }
  __syncthreads();
}

__global__ __launch_bounds__(1024)
void select_kernel(const float* __restrict__ r,
                   float* __restrict__ out_idx_f,
                   float* __restrict__ out_attn,
                   int* __restrict__ sel_idx) {
  __shared__ float sc[CAND];
  __shared__ int   id[CAND];
  __shared__ float red[1024];
  __shared__ float s_gmax, s_gsum;

  const int tid = threadIdx.x;
  const int b   = blockIdx.x;
  const float* rb = r + (size_t)b * NN;

  // init memory = first M tokens
  for (int t = tid; t < MMEM; t += 1024) { sc[t] = rb[t]; id[t] = t; }

  for (int iter = 0; iter < NITER; ++iter) {
    __syncthreads();
    int start = MMEM + iter * IITER;
    for (int t = tid; t < IITER; t += 1024) {
      sc[MMEM + t] = rb[start + t];
      id[MMEM + t] = start + t;
    }
    bitonic_sort_desc(sc, id, tid);

    if (iter == NITER - 1) {
      // pre-knockout softmax stats over all 2048 candidates
      float gmax = sc[0];
      float part = 0.f;
      for (int t = tid; t < CAND; t += 1024) part += __expf(sc[t] - gmax);
      red[tid] = part;
      __syncthreads();
      for (int s = 512; s > 0; s >>= 1) {
        if (tid < s) red[tid] += red[tid + s];
        __syncthreads();
      }
      if (tid == 0) { s_gmax = gmax; s_gsum = red[0]; }
      __syncthreads();
    }

    // random knockout of top-K (p = 0.1); hash RNG stands in for jax threefry
    if (tid < MASKK) {
      unsigned seed = 0x9E3779B9u * (unsigned)(iter + 1)
                    ^ 0x85EBCA6Bu * (unsigned)(b + 1)
                    ^ 0xC2B2AE35u * (unsigned)(tid + 1);
      if (hash01(seed) < 0.1f) sc[tid] = -3.0e38f;
    }
    bitonic_sort_desc(sc, id, tid);   // re-rank; knocked entries leave top-M
  }

  for (int t = tid; t < MMEM; t += 1024) {
    int idx = id[t];
    sel_idx[(size_t)b * MMEM + t]   = idx;
    out_idx_f[(size_t)b * MMEM + t] = (float)idx;
    out_attn[(size_t)b * MMEM + t]  = __expf(sc[t] - s_gmax) / s_gsum;
  }
}

// =====================================================================
// Kernel 3: gather winning raw patches: out[b,m,:] = patches[b,idx,:]
// =====================================================================
__global__ __launch_bounds__(128)
void gather_kernel(const float* __restrict__ patches,
                   const int* __restrict__ sel_idx,
                   float* __restrict__ outp) {
  int row = blockIdx.x;             // b*M + m
  int b   = row / MMEM;
  int src = sel_idx[row];
  const float4* s4 = (const float4*)(patches + ((size_t)b * NN + src) * CC);
  float4*       d4 = (float4*)(outp + (size_t)row * CC);
  for (int c = threadIdx.x; c < CC / 4; c += blockDim.x) d4[c] = s4[c];
}

extern "C" void kernel_launch(void* const* d_in, const int* in_sizes, int n_in,
                              void* d_out, int out_size, void* d_ws, size_t ws_size,
                              hipStream_t stream) {
  (void)in_sizes; (void)n_in; (void)out_size; (void)ws_size;
  const float* patches  = (const float*)d_in[0];
  const float* W_enc    = (const float*)d_in[1];
  const float* b_enc    = (const float*)d_in[2];
  const float* bn_gamma = (const float*)d_in[3];
  const float* bn_beta  = (const float*)d_in[4];
  const float* bn_mean  = (const float*)d_in[5];
  const float* bn_var   = (const float*)d_in[6];
  const float* w_score  = (const float*)d_in[7];

  // workspace: scores r[B*N] then selected int indices [B*M]
  float* r   = (float*)d_ws;
  int*   sel = (int*)((char*)d_ws + (size_t)BB * NN * sizeof(float));

  // output: mem_patch [B,M,C] | mem_idx [B,M] | attn_top [B,M] (flat, fp32)
  float* out_patch = (float*)d_out;
  float* out_idx   = out_patch + (size_t)BB * MMEM * CC;
  float* out_attn  = out_idx + (size_t)BB * MMEM;

  encode_score_kernel<<<dim3((BB * NN) / 128), dim3(256), 0, stream>>>(
      patches, W_enc, b_enc, bn_gamma, bn_beta, bn_mean, bn_var, w_score, r);
  select_kernel<<<dim3(BB), dim3(1024), 0, stream>>>(r, out_idx, out_attn, sel);
  gather_kernel<<<dim3(BB * MMEM), dim3(128), 0, stream>>>(patches, sel, out_patch);
}
